// NLReader_76441827934723
// MI455X (gfx1250) — matrix operand order repaired and checked by reading, not compiled, not run on hardware
//
#include <hip/hip_runtime.h>
#include <hip/hip_bf16.h>
#include <math.h>

// ---------------- problem constants ----------------
#define LL     128          // sequence length
#define BB     64           // batch
#define EE     512          // embedding dim
#define HH     512          // hidden per direction
#define NG     2048         // 4*HH gate width
#define KCAT   1024         // EE + HH concatenated K
#define NWGD   32           // workgroups per direction (each owns 16 hidden cols)
#define WSTRIDE 1032        // padded LDS row stride (bf16 elems): 2064B = +4 banks/row

typedef __attribute__((ext_vector_type(8)))  __bf16 bf16x8;
typedef __attribute__((ext_vector_type(16))) __bf16 bf16x16;
typedef __attribute__((ext_vector_type(8)))  float  floatx8;

// float -> bf16 (round to nearest even), stored as ushort
__device__ inline unsigned short f2bf(float f) {
    unsigned u = __float_as_uint(f);
    u += 0x7FFFu + ((u >> 16) & 1u);
    return (unsigned short)(u >> 16);
}

__device__ inline float sigf(float x) { return 1.0f / (1.0f + __expf(-x)); }

// A-fragment (16x32 bf16, MxK) from global: lane row = lane%16;
// half=lane/16 -> K chunks {0..7,16..23} (half 0) / {8..15,24..31} (half 1)
__device__ inline bf16x16 load_A(const __bf16* row, int kbase, int half) {
    const __bf16* p = row + kbase + half * 8;
    union { bf16x16 v; bf16x8 h[2]; } u;
    u.h[0] = *reinterpret_cast<const bf16x8*>(p);
    u.h[1] = *reinterpret_cast<const bf16x8*>(p + 16);
    return u.v;
}

// B-fragment (32x16 bf16, KxN) from LDS: lane col = lane%16;
// half=lane/16 -> K chunks {0..7,16..23} (half 0) / {8..15,24..31} (half 1)
__device__ inline bf16x16 load_B(const __bf16* row, int kbase, int half) {
    const __bf16* p = row + kbase + half * 8;
    union { bf16x16 v; bf16x8 h[2]; } u;
    u.h[0] = *reinterpret_cast<const bf16x8*>(p);
    u.h[1] = *reinterpret_cast<const bf16x8*>(p + 16);
    return u.v;
}

__device__ inline floatx8 wmma_bf16(bf16x16 a, bf16x16 b, floatx8 c) {
    return __builtin_amdgcn_wmma_f32_16x16x32_bf16(
        /*neg_a=*/false, a, /*neg_b=*/false, b,
        /*c_mod=*/(short)0, c, /*reuse_a=*/false, /*reuse_b=*/false);
}

// device-wide barrier among the NWGD workgroups of one direction
__device__ inline void grid_barrier(unsigned* cnt, unsigned target) {
    __threadfence();
    __syncthreads();
    if (threadIdx.x == 0) {
        __hip_atomic_fetch_add(cnt, 1u, __ATOMIC_ACQ_REL, __HIP_MEMORY_SCOPE_AGENT);
        while (__hip_atomic_load(cnt, __ATOMIC_ACQUIRE, __HIP_MEMORY_SCOPE_AGENT) < target) {
            __builtin_amdgcn_s_sleep(1);
        }
    }
    __syncthreads();
    __builtin_amdgcn_fence(__ATOMIC_ACQUIRE, "agent");
}

// ---------------- prep kernels ----------------

// embedding gather + bf16 convert; padding (data == -1) -> 0
__global__ void k_embed(const int* __restrict__ data,
                        const float* __restrict__ table,
                        unsigned short* __restrict__ out) {
    int idx = blockIdx.x * blockDim.x + threadIdx.x;
    if (idx >= LL * BB * EE) return;
    int tb = idx >> 9;        // / EE
    int e  = idx & (EE - 1);
    int d  = data[tb];
    float v = (d < 0) ? 0.0f : table[(size_t)d * EE + e];
    out[idx] = f2bf(v);
}

// Wcat[dir][n][k] = bf16( k<EE ? Wih[n][k] : Whh[n][k-EE] )
__global__ void k_weights(const float* __restrict__ WihF, const float* __restrict__ WhhF,
                          const float* __restrict__ WihB, const float* __restrict__ WhhB,
                          unsigned short* __restrict__ Wcat) {
    int idx = blockIdx.x * blockDim.x + threadIdx.x;
    if (idx >= 2 * NG * KCAT) return;
    int dir = idx / (NG * KCAT);
    int rem = idx - dir * (NG * KCAT);
    int r = rem / KCAT;
    int k = rem - r * KCAT;
    const float* Wih = dir ? WihB : WihF;
    const float* Whh = dir ? WhhB : WhhF;
    float v = (k < EE) ? Wih[(size_t)r * EE + k] : Whh[(size_t)r * HH + (k - EE)];
    Wcat[idx] = f2bf(v);
}

__global__ void k_bias(const float* __restrict__ bihF, const float* __restrict__ bhhF,
                       const float* __restrict__ bihB, const float* __restrict__ bhhB,
                       float* __restrict__ bias) {
    int idx = blockIdx.x * blockDim.x + threadIdx.x;
    if (idx >= 2 * NG) return;
    int dir = idx / NG;
    int n   = idx - dir * NG;
    bias[idx] = dir ? (bihB[n] + bhhB[n]) : (bihF[n] + bhhF[n]);
}

// zero h double-buffers (2 dirs x 2 bufs x 64x512 bf16) and barrier counters
__global__ void k_init(unsigned short* __restrict__ hbuf, unsigned* __restrict__ bar) {
    int idx = blockIdx.x * blockDim.x + threadIdx.x;
    if (idx < 2 * 2 * BB * HH) hbuf[idx] = 0;
    if (idx < 2) bar[idx * 64] = 0u;
}

// ---------------- persistent bi-LSTM kernel ----------------
// grid = 64 WGs x 256 threads. blocks [0,32): forward, [32,64): backward.
// Each WG owns 16 hidden columns -> cell state c stays in registers; the WG's
// 64x1024 bf16 weight slice is staged in LDS once and reused for all 128 steps.
__global__ void __launch_bounds__(256, 1)
k_lstm(const unsigned short* __restrict__ emb_us,
       const unsigned short* __restrict__ Wcat_us,
       const float* __restrict__ bias,
       const float* __restrict__ mask,
       unsigned short* __restrict__ hbuf_us,
       unsigned* __restrict__ bar,
       float* __restrict__ out) {
    const int wg      = blockIdx.x;
    const int dir     = wg >> 5;
    const int colbase = (wg & 31) * 16;

    const __bf16* emb = reinterpret_cast<const __bf16*>(emb_us);
    const __bf16* W   = reinterpret_cast<const __bf16*>(Wcat_us) + (size_t)dir * NG * KCAT;
    const float*  bs  = bias + dir * NG;
    __bf16*       hb  = reinterpret_cast<__bf16*>(hbuf_us) + (size_t)dir * 2 * BB * HH;
    unsigned*     cnt = bar + dir * 64;

    const int tid   = threadIdx.x;
    const int lane  = tid & 31;
    const int wave  = tid >> 5;
    const int mt    = wave & 3;            // batch tile (rows mt*16..mt*16+15)
    const int g0    = (wave >> 2) << 1;    // this wave handles gates g0, g0+1
    const int idx16 = lane & 15;
    const int half  = lane >> 4;

    // LDS: 64 weight rows (4 gates x 16 cols) x KCAT, padded stride -> ~129 KB
    __shared__ __bf16 wlds[64 * WSTRIDE];
    __shared__ float  gbuf[4][BB][17];     // padded gate staging, ~17.4 KB

    // ---- stage this WG's weight slice into LDS (once) ----
    for (int i = tid; i < 64 * (KCAT / 8); i += 256) {
        int r  = i >> 7;                   // LDS row 0..63
        int c  = (i & 127) << 3;           // elem offset within row
        int gr = (r >> 4) * HH + colbase + (r & 15);   // global gate row
        *reinterpret_cast<bf16x8*>(&wlds[r * WSTRIDE + c]) =
            *reinterpret_cast<const bf16x8*>(W + (size_t)gr * KCAT + c);
    }
    __syncthreads();

    const __bf16* Brow0 = &wlds[(g0 * 16 + idx16) * WSTRIDE];
    const __bf16* Brow1 = &wlds[((g0 + 1) * 16 + idx16) * WSTRIDE];

    // elementwise phase mapping: thread -> (batch row, 4 columns)
    const int erow  = tid >> 2;            // 0..63
    const int ecol0 = (tid & 3) << 2;      // 0,4,8,12
    float cst[4] = {0.0f, 0.0f, 0.0f, 0.0f};

    for (int step = 0; step < LL; ++step) {
        // keep loads inside the sequential loop (prevents hoisting -> spills)
        asm volatile("" ::: "memory");

        const int tt = dir ? (LL - 1 - step) : step;
        const __bf16* xr = emb + ((size_t)tt * BB + mt * 16 + idx16) * EE;
        const __bf16* hr = hb + (size_t)(step & 1) * BB * HH
                              + (size_t)(mt * 16 + idx16) * HH;

        if (step + 1 < LL) {  // prefetch next step's x rows (global_prefetch_b8)
            int tn = dir ? (tt - 1) : (tt + 1);
            __builtin_prefetch(emb_us + ((size_t)tn * BB + mt * 16 + idx16) * EE, 0, 1);
        }

        floatx8 a0 = {0.f,0.f,0.f,0.f,0.f,0.f,0.f,0.f};
        floatx8 a1 = {0.f,0.f,0.f,0.f,0.f,0.f,0.f,0.f};

        #pragma unroll
        for (int kk = 0; kk < 16; ++kk) {          // K = 0..511 : x @ Wih^T
            int kb = kk << 5;
            bf16x16 a = load_A(xr, kb, half);
            a0 = wmma_bf16(a, load_B(Brow0, kb, half), a0);
            a1 = wmma_bf16(a, load_B(Brow1, kb, half), a1);
        }
        #pragma unroll
        for (int kk = 0; kk < 16; ++kk) {          // K = 512..1023 : h @ Whh^T
            int kb = kk << 5;
            bf16x16 a = load_A(hr, kb, half);
            a0 = wmma_bf16(a, load_B(Brow0, EE + kb, half), a0);
            a1 = wmma_bf16(a, load_B(Brow1, EE + kb, half), a1);
        }

        // C layout: VGPR r -> row r + half*8, col = lane%16
        #pragma unroll
        for (int r = 0; r < 8; ++r) {
            gbuf[g0    ][mt * 16 + half * 8 + r][idx16] = a0[r];
            gbuf[g0 + 1][mt * 16 + half * 8 + r][idx16] = a1[r];
        }
        __syncthreads();

        // ---- elementwise LSTM cell update for owned (row, col) elements ----
        const float m = mask[tt * BB + erow];
        __bf16* hw = hb + (size_t)((step + 1) & 1) * BB * HH
                        + (size_t)erow * HH + colbase;
        float* orow = out + ((size_t)tt * BB + erow) * (2 * HH) + dir * HH + colbase;

        #pragma unroll
        for (int e = 0; e < 4; ++e) {
            int col = ecol0 + e;
            int j   = colbase + col;
            float gi = gbuf[0][erow][col] + bs[j];
            float gf = gbuf[1][erow][col] + bs[HH + j];
            float gg = gbuf[2][erow][col] + bs[2 * HH + j];
            float go = gbuf[3][erow][col] + bs[3 * HH + j];
            float cn = sigf(gf) * cst[e] + sigf(gi) * tanhf(gg);
            float hn = sigf(go) * tanhf(cn);
            hn *= m;
            cn *= m;
            cst[e] = cn;
            orow[col] = hn;
            reinterpret_cast<unsigned short*>(hw)[col] = f2bf(hn);
        }

        if (step != LL - 1) {
            grid_barrier(cnt, (unsigned)(NWGD * (step + 1)));
        }
    }
}

// ---------------- launch ----------------
extern "C" void kernel_launch(void* const* d_in, const int* in_sizes, int n_in,
                              void* d_out, int out_size, void* d_ws, size_t ws_size,
                              hipStream_t stream) {
    const int*   data  = (const int*)  d_in[0];
    const float* mask  = (const float*)d_in[1];
    const float* table = (const float*)d_in[2];
    const float* WihF  = (const float*)d_in[3];
    const float* WhhF  = (const float*)d_in[4];
    const float* bihF  = (const float*)d_in[5];
    const float* bhhF  = (const float*)d_in[6];
    const float* WihB  = (const float*)d_in[7];
    const float* WhhB  = (const float*)d_in[8];
    const float* bihB  = (const float*)d_in[9];
    const float* bhhB  = (const float*)d_in[10];
    float* out = (float*)d_out;

    // workspace layout (all 256B-aligned)
    char* ws = (char*)d_ws;
    const size_t WCAT_BYTES = (size_t)2 * NG * KCAT * 2;   // 8 MiB
    const size_t BIAS_BYTES = (size_t)2 * NG * 4;          // 16 KiB
    const size_t EMB_BYTES  = (size_t)LL * BB * EE * 2;    // 8 MiB
    const size_t H_BYTES    = (size_t)2 * 2 * BB * HH * 2; // 128 KiB
    const size_t OFF_WCAT = 0;
    const size_t OFF_BIAS = OFF_WCAT + WCAT_BYTES;
    const size_t OFF_EMB  = OFF_BIAS + BIAS_BYTES;
    const size_t OFF_H    = OFF_EMB + EMB_BYTES;
    const size_t OFF_BAR  = OFF_H + H_BYTES;

    unsigned short* Wcat = (unsigned short*)(ws + OFF_WCAT);
    float*          bias = (float*)         (ws + OFF_BIAS);
    unsigned short* embb = (unsigned short*)(ws + OFF_EMB);
    unsigned short* hbuf = (unsigned short*)(ws + OFF_H);
    unsigned*       bar  = (unsigned*)      (ws + OFF_BAR);

    {   // weight concat + bf16 convert
        int n = 2 * NG * KCAT;
        k_weights<<<(n + 255) / 256, 256, 0, stream>>>(WihF, WhhF, WihB, WhhB, Wcat);
    }
    {   // bias sum
        int n = 2 * NG;
        k_bias<<<(n + 255) / 256, 256, 0, stream>>>(bihF, bhhF, bihB, bhhB, bias);
    }
    {   // embedding gather
        int n = LL * BB * EE;
        k_embed<<<(n + 255) / 256, 256, 0, stream>>>(data, table, embb);
    }
    {   // zero h buffers + reset barrier counters (every launch; ws is poisoned)
        int n = 2 * 2 * BB * HH;
        k_init<<<(n + 255) / 256, 256, 0, stream>>>(hbuf, bar);
    }
    // persistent bidirectional LSTM: 32 WGs forward + 32 WGs backward
    k_lstm<<<2 * NWGD, 256, 0, stream>>>(embb, Wcat, bias, mask, hbuf, bar, out);

    (void)in_sizes; (void)n_in; (void)out_size; (void)ws_size;
}
